// MoELayer_51445118271939
// MI455X (gfx1250) — compile-verified, matrix operand order checked by reading, not run on hardware
//
#include <hip/hip_runtime.h>
#include <hip/hip_bf16.h>
#include <math.h>

// ---------------------------------------------------------------------------
// MoE (dim=1024, hidden=4096, 8 experts, top-2, swiglu) for MI455X (gfx1250).
// Roofline: ~413 GFLOP vs ~436 MB fp32 weight traffic -> memory-bound
// (~19 us at 23.3 TB/s). Strategy: gathered dispatch (weights read once,
// expert tiles hit the 192 MB L2), fp32->bf16 convert in registers, all GEMM
// work via v_wmma_f32_16x16x32_bf16 with f32 accumulation.
// ---------------------------------------------------------------------------

#define DIM    1024
#define HIDDEN 4096
#define TWOH   8192
#define NEXP   8
#define NTOK   8192   // 4 * 2048 tokens
#define CAP    8192   // max tokens per expert (top-2 experts are distinct)

typedef __attribute__((ext_vector_type(16))) __bf16 bf16x16;
typedef __attribute__((ext_vector_type(8)))  __bf16 bf16x8;
typedef __attribute__((ext_vector_type(4)))  __bf16 bf16x4;
typedef __attribute__((ext_vector_type(8)))  float  f32x8;

// ---- WMMA fragment helpers -------------------------------------------------
// A (16x32 bf16) per ISA 7.12.2: lane = h*16 + row; element j holds
// k = kb + (j>>3)*16 + h*8 + (j&7)  => two contiguous 8-element runs.
// B (32x16) symmetric: lane holds column N = lane&15, same k swizzle.

__device__ __forceinline__ bf16x16 frag_lds(const __bf16* rowbase, int kb, int h) {
  bf16x8 lo = *(const bf16x8*)(rowbase + kb + h * 8);
  bf16x8 hi = *(const bf16x8*)(rowbase + kb + 16 + h * 8);
  return __builtin_shufflevector(lo, hi, 0,1,2,3,4,5,6,7,8,9,10,11,12,13,14,15);
}

__device__ __forceinline__ bf16x8 cvt8f(const float* p) {
  const float4 a = *(const float4*)p;
  const float4 b = *(const float4*)(p + 4);
  bf16x8 r;
  r[0] = (__bf16)a.x; r[1] = (__bf16)a.y; r[2] = (__bf16)a.z; r[3] = (__bf16)a.w;
  r[4] = (__bf16)b.x; r[5] = (__bf16)b.y; r[6] = (__bf16)b.z; r[7] = (__bf16)b.w;
  return r;
}

// Build a bf16 fragment from an fp32 row in global memory (weights).
__device__ __forceinline__ bf16x16 frag_gf(const float* rowbase, int kb, int h) {
  bf16x8 lo = cvt8f(rowbase + kb + h * 8);
  bf16x8 hi = cvt8f(rowbase + kb + 16 + h * 8);
  return __builtin_shufflevector(lo, hi, 0,1,2,3,4,5,6,7,8,9,10,11,12,13,14,15);
}

// ---- Kernel 1: router (logits -> top2 -> softmax -> per-expert lists) ------
__global__ void router_kernel(const float* __restrict__ x,
                              const float* __restrict__ rw,
                              const float* __restrict__ rb,
                              int*   __restrict__ counts,
                              int*   __restrict__ lists,
                              float* __restrict__ gates) {
  const int tok  = (blockIdx.x * blockDim.x + threadIdx.x) >> 5;
  const int lane = threadIdx.x & 31;
  if (tok >= NTOK) return;

  const float* xr = x + (size_t)tok * DIM;
  float acc[NEXP];
#pragma unroll
  for (int e = 0; e < NEXP; ++e) acc[e] = 0.f;

  for (int d = lane; d < DIM; d += 32) {
    const float xv = xr[d];
#pragma unroll
    for (int e = 0; e < NEXP; ++e) acc[e] += xv * rw[e * DIM + d];
  }
#pragma unroll
  for (int e = 0; e < NEXP; ++e)
    for (int off = 16; off > 0; off >>= 1)
      acc[e] += __shfl_xor(acc[e], off, 32);

  if (lane == 0) {
    float l[NEXP];
#pragma unroll
    for (int e = 0; e < NEXP; ++e) l[e] = acc[e] + rb[e];   // TEMP == 1.0

    int e0 = 0; float l0 = l[0];
#pragma unroll
    for (int e = 1; e < NEXP; ++e) if (l[e] > l0) { l0 = l[e]; e0 = e; }
    int e1 = (e0 == 0) ? 1 : 0; float l1 = l[e1];
#pragma unroll
    for (int e = 0; e < NEXP; ++e)
      if (e != e0 && l[e] > l1) { l1 = l[e]; e1 = e; }

    const float p   = __expf(l1 - l0);
    const float inv = 1.f / (1.f + p);

    int p0 = atomicAdd(&counts[e0], 1);
    lists[e0 * CAP + p0] = tok;  gates[e0 * CAP + p0] = inv;
    int p1 = atomicAdd(&counts[e1], 1);
    lists[e1 * CAP + p1] = tok;  gates[e1 * CAP + p1] = p * inv;
  }
}

// ---- Kernel 2: fused gathered swiglu FFN (fc1 + silu*gate + fc2 + combine) -
// Block = 4 waves = 128 threads, owns 16 tokens of one expert.
// Wave w computes hidden cols [chunk+16w, chunk+16w+16) in fc1, and output
// cols [256w, 256w+256) in fc2 (16 persistent f32x8 accumulators).
__global__ void __launch_bounds__(128)
moe_ffn_kernel(const float* __restrict__ w1, const float* __restrict__ b1,
               const float* __restrict__ w2, const float* __restrict__ b2,
               const float* __restrict__ x,
               const int* __restrict__ counts, const int* __restrict__ lists,
               const float* __restrict__ gates,
               float* __restrict__ out) {
  const int e   = blockIdx.y;
  const int cnt = counts[e];
  const int m0  = blockIdx.x * 16;
  if (m0 >= cnt) return;

  __shared__ __align__(16) __bf16 xs[16 * 1032];  // 16 tokens x 1024 (pad 8)
  __shared__ __align__(16) __bf16 hs[16 * 72];    // 16 tokens x 64   (pad 8)
  __shared__ int   stok[16];
  __shared__ float sgate[16];

  const int tid = threadIdx.x;
  const int lane = tid & 31;
  const int wv = tid >> 5;        // wave 0..3
  const int hh = lane >> 4;       // lane half (K-group select)
  const int ln = lane & 15;       // row/col within tile

  if (tid < 16) {
    const int ri = m0 + tid;
    const int ci = (ri < cnt) ? ri : (cnt - 1);
    stok[tid]  = lists[e * CAP + ci];
    sgate[tid] = (ri < cnt) ? gates[e * CAP + ci] : 0.f;
  }
  __syncthreads();

  // Stage the 16x1024 token tile into LDS as bf16 (padded rows: no bank
  // conflicts on the 16B A-fragment reads).
  for (int i = tid; i < 16 * 256; i += 128) {
    const int row = i >> 8, c4 = i & 255;
    const float4 f = *(const float4*)(x + (size_t)stok[row] * DIM + c4 * 4);
    bf16x4 b;
    b[0] = (__bf16)f.x; b[1] = (__bf16)f.y; b[2] = (__bf16)f.z; b[3] = (__bf16)f.w;
    *(bf16x4*)(xs + row * 1032 + c4 * 4) = b;
  }
  __syncthreads();

  f32x8 acc2[16];
#pragma unroll
  for (int i = 0; i < 16; ++i)
#pragma unroll
    for (int r = 0; r < 8; ++r) acc2[i][r] = 0.f;

  const float* w1g = w1 + (size_t)e * TWOH * DIM;
  const float* w2g = w2 + (size_t)e * DIM * HIDDEN;

  for (int hc = 0; hc < HIDDEN / 64; ++hc) {
    // ---- fc1: this wave's 16 hidden cols (gate half + value half) ----
    const int n1 = hc * 64 + wv * 16;
    const float* rowg = w1g + (size_t)(n1 + ln) * DIM;           // h1 rows
    const float* rowv = rowg + (size_t)HIDDEN * DIM;             // h2 rows
    f32x8 cg, cv;
#pragma unroll
    for (int r = 0; r < 8; ++r) { cg[r] = 0.f; cv[r] = 0.f; }

#pragma unroll 4
    for (int kt = 0; kt < DIM / 32; ++kt) {
      const int kb = kt * 32;
      __builtin_prefetch(rowg + kb + 64, 0, 3);   // global_prefetch, WGP scope
      __builtin_prefetch(rowv + kb + 64, 0, 3);
      bf16x16 a  = frag_lds(xs + ln * 1032, kb, hh);
      bf16x16 bg = frag_gf(rowg, kb, hh);
      bf16x16 bv = frag_gf(rowv, kb, hh);
      cg = __builtin_amdgcn_wmma_f32_16x16x32_bf16(false, a, false, bg,
                                                   (short)0, cg, false, false);
      cv = __builtin_amdgcn_wmma_f32_16x16x32_bf16(false, a, false, bv,
                                                   (short)0, cv, false, false);
    }

    // swiglu epilogue -> h chunk in LDS (bf16)
    const float bgs = b1[(size_t)e * TWOH + n1 + ln];
    const float bvs = b1[(size_t)e * TWOH + n1 + ln + HIDDEN];
#pragma unroll
    for (int r = 0; r < 8; ++r) {
      const int M = r + 8 * hh;                 // token row (C layout)
      const float g = cg[r] + bgs;
      const float v = cv[r] + bvs;
      const float hval = (g / (1.f + __expf(-g))) * v;   // silu(g) * v
      hs[M * 72 + wv * 16 + ln] = (__bf16)hval;
    }
    __syncthreads();

    // ---- fc2 partial: out[16 x 1024] += h_chunk[16 x 64] @ w2[:,chunk]^T ---
#pragma unroll 1
    for (int nt = 0; nt < 16; ++nt) {
      const int n2 = wv * 256 + nt * 16 + ln;
      const float* roww = w2g + (size_t)n2 * HIDDEN + hc * 64;
      __builtin_prefetch(roww + 64, 0, 3);      // next chunk's slice, same row
#pragma unroll
      for (int kk = 0; kk < 2; ++kk) {
        bf16x16 ah = frag_lds(hs + ln * 72, kk * 32, hh);
        bf16x16 bw = frag_gf(roww, kk * 32, hh);
        acc2[nt] = __builtin_amdgcn_wmma_f32_16x16x32_bf16(false, ah, false, bw,
                                                           (short)0, acc2[nt],
                                                           false, false);
      }
    }
    __syncthreads();   // protect hs before next chunk overwrites it
  }

  // ---- gate-weighted scatter (each token gets exactly K=2 contributions) ---
  // Native f32 atomics (GLOBAL_ATOMIC_ADD_F32), relaxed, device scope.
#pragma unroll 1
  for (int nt = 0; nt < 16; ++nt) {
    const int col = wv * 256 + nt * 16 + ln;
    const float bb = b2[(size_t)e * DIM + col];
#pragma unroll
    for (int r = 0; r < 8; ++r) {
      const int M = r + 8 * hh;
      const float gte = sgate[M];
      if (gte != 0.f) {
        float* dst = out + (size_t)stok[M] * DIM + col;
        __hip_atomic_fetch_add(dst, gte * (acc2[nt][r] + bb),
                               __ATOMIC_RELAXED, __HIP_MEMORY_SCOPE_AGENT);
      }
    }
  }
}

// ---------------------------------------------------------------------------
extern "C" void kernel_launch(void* const* d_in, const int* in_sizes, int n_in,
                              void* d_out, int out_size, void* d_ws, size_t ws_size,
                              hipStream_t stream) {
  const float* x  = (const float*)d_in[0];
  const float* rw = (const float*)d_in[1];
  const float* rb = (const float*)d_in[2];
  const float* w1 = (const float*)d_in[3];
  const float* b1 = (const float*)d_in[4];
  const float* w2 = (const float*)d_in[5];
  const float* b2 = (const float*)d_in[6];
  float* out = (float*)d_out;

  // workspace: [counts:256B][lists: 8*8192 int][gates: 8*8192 float] ~ 0.5 MB
  int*   counts = (int*)d_ws;
  int*   lists  = (int*)((char*)d_ws + 256);
  float* gates  = (float*)((char*)d_ws + 256 + (size_t)NEXP * CAP * sizeof(int));

  hipMemsetAsync(d_ws, 0, 256, stream);                              // counts = 0
  hipMemsetAsync(d_out, 0, (size_t)out_size * sizeof(float), stream);

  router_kernel<<<NTOK / 8, 256, 0, stream>>>(x, rw, rb, counts, lists, gates);

  dim3 grid(CAP / 16, NEXP);
  moe_ffn_kernel<<<grid, 128, 0, stream>>>(w1, b1, w2, b2, x,
                                           counts, lists, gates, out);
}
